// C_T_F_Attention_90529320665770
// MI455X (gfx1250) — matrix operand order repaired and checked by reading, not compiled, not run on hardware
//
#include <hip/hip_runtime.h>
#include <hip/hip_bf16.h>

// ---- CDNA5 vector types (POD ext-vectors so they can live in unions) ----
typedef __attribute__((ext_vector_type(16))) __bf16        v16bf;
typedef __attribute__((ext_vector_type(8)))  float         v8f;
typedef __attribute__((ext_vector_type(4)))  unsigned int  u32x4;
typedef __attribute__((ext_vector_type(4)))  float         f32x4;

#define EPSF 1e-5f
constexpr int B_  = 32, C_ = 256, FR_ = 7, T_ = 251;
constexpr int P_  = FR_ * T_;     // 1757 (flattened fr,t — contiguous in memory)
constexpr int NF  = B_ * FR_;     // 224
constexpr int NT  = B_ * T_;      // 8032

// ---- workspace layout (float offsets); total ~1.51 MB ----
constexpr int WS_QCE = 0,   WS_KCE = 256, WS_QFE = 512, WS_KFE = 768;
constexpr int WS_QTE = 1024, WS_KTE = 1280;
constexpr int WS_SCAL = 1536;             // [0..5]=tbar q/k c,f,t  [8..10]=softmax scales
constexpr int WS_SV = 1552, WS_TV2 = 1808;
constexpr int WS_DC = 2064, WS_DF = 2320, WS_DT = 2576; // dt uses 7 (pad 8)
constexpr int WS_QVC = 2592;
constexpr int WS_KVC = WS_QVC + NF * 256;
constexpr int WS_QVF = WS_KVC + NF * 256;
constexpr int WS_KVF = WS_QVF + NF * 256;
constexpr int WS_QVT = WS_KVF + NF * 256;
constexpr int WS_KVT = WS_QVT + NT * 8;
constexpr int WS_ASWZ = WS_KVT + NT * 8;  // 32768 uint32 (bf16-pair swizzled v_w)

__device__ __forceinline__ unsigned int f2bf(float f) {
  unsigned int u = __float_as_uint(f);
  return (u + 0x7FFFu + ((u >> 16) & 1u)) >> 16;  // RNE f32 -> bf16
}

// ---------------------------------------------------------------------------
// Prep 1: effective contraction vectors eff[c] = sum_o s[o]*W[o,c] and
// tbar = mean_o(beta[o] - mean[o]*s[o]) for the 6 (q,k)x(c,f,t) pairs.
// ---------------------------------------------------------------------------
__global__ void __launch_bounds__(256) prep_eff(
    const float* qc_w, const float* qc_bn, const float* kc_w, const float* kc_bn,
    const float* qf_w, const float* qf_bn, const float* kf_w, const float* kf_bn,
    const float* qt_w, const float* qt_bn, const float* kt_w, const float* kt_bn,
    float* ws) {
  __shared__ float s[256], tl[256];
  const int bid = blockIdx.x, tid = threadIdx.x;
  const float* w; const float* bn; int O, Cd; float* eff;
  switch (bid) {
    case 0:  w = qc_w; bn = qc_bn; O = T_; Cd = T_; eff = ws + WS_QCE; break;
    case 1:  w = kc_w; bn = kc_bn; O = T_; Cd = T_; eff = ws + WS_KCE; break;
    case 2:  w = qf_w; bn = qf_bn; O = C_; Cd = C_; eff = ws + WS_QFE; break;
    case 3:  w = kf_w; bn = kf_bn; O = C_; Cd = C_; eff = ws + WS_KFE; break;
    case 4:  w = qt_w; bn = qt_bn; O = C_; Cd = C_; eff = ws + WS_QTE; break;
    default: w = kt_w; bn = kt_bn; O = C_; Cd = C_; eff = ws + WS_KTE; break;
  }
  if (tid < O) {
    float g = bn[tid], be = bn[O + tid], mn = bn[2 * O + tid], va = bn[3 * O + tid];
    float sv = g * rsqrtf(va + EPSF);
    s[tid] = sv; tl[tid] = be - mn * sv;
  } else { s[tid] = 0.f; tl[tid] = 0.f; }
  __syncthreads();
  if (tid < Cd) {
    float acc = 0.f;
    for (int o = 0; o < O; ++o) acc += s[o] * w[o * Cd + tid];
    eff[tid] = acc;
  } else {
    eff[tid] = 0.f;
  }
  if (tid == 0) {
    float tb = 0.f;
    for (int o = 0; o < O; ++o) tb += tl[o];
    ws[WS_SCAL + bid] = tb / (float)O;
  }
}

// ---------------------------------------------------------------------------
// Prep 2: swizzle v_w into WMMA bf16 A-fragment layout (ISA 7.12.2):
//   lane = M (0..15, halves), VGPR i holds K pair; i<4 -> k0=2i+8*half,
//   i>=4 -> k0=16+2(i-4)+8*half.  Also fold v-BN and zero d-accumulators.
// ---------------------------------------------------------------------------
__global__ void __launch_bounds__(256) prep_misc(
    const float* v_w, const float* v_b, const float* v_bn,
    const float* lc_bn, const float* lf_bn, const float* lt_bn, float* ws) {
  const int j = blockIdx.x * 256 + threadIdx.x;
  unsigned int* Aswz = (unsigned int*)(ws + WS_ASWZ);
  if (j < 32768) {
    int mtile = j >> 11;
    int rem   = j & 2047;
    int ks    = rem >> 8;
    int rem2  = rem & 255;
    int lane  = rem2 >> 3;
    int i     = rem2 & 7;
    int half  = lane >> 4;
    int m     = mtile * 16 + (lane & 15);
    int k     = ks * 32 + ((i < 4) ? (2 * i + 8 * half) : (16 + 2 * (i - 4) + 8 * half));
    unsigned lo = f2bf(v_w[m * C_ + k]);
    unsigned hi = f2bf(v_w[m * C_ + k + 1]);
    Aswz[j] = lo | (hi << 16);
  } else if (j < 33024) {
    int o = j - 32768;
    float g = v_bn[o], be = v_bn[C_ + o], mn = v_bn[2 * C_ + o], va = v_bn[3 * C_ + o];
    float sv = g * rsqrtf(va + EPSF);
    ws[WS_SV + o]  = sv;
    ws[WS_TV2 + o] = v_b[o] * sv + (be - mn * sv);   // folds +v_b into BN shift
  } else if (j < 33280) ws[WS_DC + j - 33024] = 0.f;
  else if (j < 33536)   ws[WS_DF + j - 33280] = 0.f;
  else if (j < 33544)   ws[WS_DT + j - 33536] = 0.f;
  else if (j == 33544)  ws[WS_SCAL + 8]  = lc_bn[0] * rsqrtf(lc_bn[3] + EPSF);
  else if (j == 33545)  ws[WS_SCAL + 9]  = lf_bn[0] * rsqrtf(lf_bn[3] + EPSF);
  else if (j == 33546)  ws[WS_SCAL + 10] = lt_bn[0] * rsqrtf(lt_bn[3] + EPSF);
}

// ---------------------------------------------------------------------------
// One pass over x per (b,fr) slab: all six qv/kv weighted reductions.
// ---------------------------------------------------------------------------
__global__ void __launch_bounds__(256) qv_pass(const float* __restrict__ x,
                                               float* __restrict__ ws) {
  __shared__ float eQf[256], eKf[256], eQt[256], eKt[256], eQc[256], eKc[256];
  const int b = blockIdx.x / FR_, fr = blockIdx.x % FR_;
  const int tid = threadIdx.x;
  eQf[tid] = ws[WS_QFE + tid]; eKf[tid] = ws[WS_KFE + tid];
  eQt[tid] = ws[WS_QTE + tid]; eKt[tid] = ws[WS_KTE + tid];
  eQc[tid] = ws[WS_QCE + tid]; eKc[tid] = ws[WS_KCE + tid];
  __syncthreads();
  const float* xb = x + (size_t)b * C_ * P_ + fr * T_;
  const int n = b * FR_ + fr;
  if (tid < T_) {  // per-t accumulators, coalesced in t
    float aQf = 0.f, aKf = 0.f, aQt = 0.f, aKt = 0.f;
    for (int ch = 0; ch < C_; ++ch) {
      float xv = xb[ch * P_ + tid];
      aQf += eQf[ch] * xv; aKf += eKf[ch] * xv;
      aQt += eQt[ch] * xv; aKt += eKt[ch] * xv;
    }
    ws[WS_QVF + n * 256 + tid] = aQf * (1.f / C_) + ws[WS_SCAL + 2];
    ws[WS_KVF + n * 256 + tid] = aKf * (1.f / C_) + ws[WS_SCAL + 3];
    const int nt = b * T_ + tid;
    ws[WS_QVT + nt * 8 + fr] = aQt * (1.f / C_) + ws[WS_SCAL + 4];
    ws[WS_KVT + nt * 8 + fr] = aKt * (1.f / C_) + ws[WS_SCAL + 5];
  }
  {  // per-channel accumulators (slab is L2-hot after phase 1)
    float aQc = 0.f, aKc = 0.f;
    const float* row = xb + tid * P_;
    for (int t = 0; t < T_; ++t) {
      float xv = row[t];
      aQc += eQc[t] * xv; aKc += eKc[t] * xv;
    }
    ws[WS_QVC + n * 256 + tid] = aQc * (1.f / T_) + ws[WS_SCAL + 0];
    ws[WS_KVC + n * 256 + tid] = aKc * (1.f / T_) + ws[WS_SCAL + 1];
  }
}

// ---------------------------------------------------------------------------
// d[i] = sum_n exp(a_i kv_i)/sum_j exp(a_i kv_j),  a_i = scale*qv_i.
// (The constant BN shift cancels inside softmax.)
// ---------------------------------------------------------------------------
__global__ void __launch_bounds__(256) diag_cf(const float* __restrict__ qv,
                                               const float* __restrict__ kv,
                                               int L, const float* scalePtr,
                                               float* dOut) {
  __shared__ float kvs[256];
  const int n = blockIdx.x, tid = threadIdx.x;
  const float scale = *scalePtr;
  kvs[tid] = (tid < L) ? kv[n * 256 + tid] : 0.f;
  __syncthreads();
  if (tid < L) {
    float a = scale * qv[n * 256 + tid];
    float denom = 0.f;
    for (int j = 0; j < L; ++j) denom += expf(a * kvs[j]);
    atomicAdd(&dOut[tid], expf(a * kvs[tid]) / denom);
  }
}

__global__ void __launch_bounds__(256) diag_t(const float* __restrict__ ws,
                                              float* __restrict__ dt) {
  __shared__ float red[256];
  const int n = blockIdx.x * 256 + threadIdx.x;
  const float scale = ws[WS_SCAL + 10];
  float acc[7];
  if (n < NT) {
    float q[7], k[7];
    for (int f = 0; f < 7; ++f) { q[f] = ws[WS_QVT + n * 8 + f]; k[f] = ws[WS_KVT + n * 8 + f]; }
    for (int i = 0; i < 7; ++i) {
      float a = scale * q[i];
      float denom = 0.f;
      for (int j = 0; j < 7; ++j) denom += expf(a * k[j]);
      acc[i] = expf(a * k[i]) / denom;
    }
  } else {
    for (int i = 0; i < 7; ++i) acc[i] = 0.f;
  }
  for (int f = 0; f < 7; ++f) {
    red[threadIdx.x] = acc[f];
    __syncthreads();
    for (int s = 128; s > 0; s >>= 1) {
      if (threadIdx.x < (unsigned)s) red[threadIdx.x] += red[threadIdx.x + s];
      __syncthreads();
    }
    if (threadIdx.x == 0) atomicAdd(&dt[f], red[0]);
    __syncthreads();
  }
}

// ---------------------------------------------------------------------------
// Main: out[b,o,p] = (Wx*s[o] + t2[o]) * (dc[o] + dt[p/T] + df[p%T]) + x.
// Per block: one b, 64 columns of p. 8 waves, each 32(M)x64(N) tile,
// K=256 in 8 steps of v_wmma_f32_16x16x32_bf16 (64 WMMAs/wave).
// ---------------------------------------------------------------------------
__global__ void __launch_bounds__(256) main_gemm(const float* __restrict__ x,
                                                 const float* __restrict__ ws,
                                                 float* __restrict__ out) {
  __shared__ unsigned int ldsB[256 * 36];  // bf16 pairs, row stride 132 dwords
  __shared__ f32x4 ep[256];                // {s, t2, dc, 0} per output channel
  __shared__ float ddt[64];                // dt[p/T]+df[p%T] per column
  const int b = blockIdx.y;
  const int pBase = blockIdx.x * 64;
  const int tid = threadIdx.x;
  const unsigned int* Aswz = (const unsigned int*)(ws + WS_ASWZ);

  { f32x4 e; e.x = ws[WS_SV + tid]; e.y = ws[WS_TV2 + tid];
    e.z = ws[WS_DC + tid]; e.w = 0.f; ep[tid] = e; }
  if (tid < 64) {
    int p = pBase + tid;
    ddt[tid] = (p < P_) ? (ws[WS_DT + p / T_] + ws[WS_DF + p % T_]) : 0.f;
  }
  // stage B tile (K=256 x N=64) as bf16 transposed: row = N column (stride 132
  // dwords), dword kp = {K=2kp, K=2kp+1} pair.
  const float* xb = x + (size_t)b * C_ * P_;
  const int cn = tid & 63, r0 = tid >> 6;
  for (int pass = 0; pass < 32; ++pass) {
    int kp = pass * 4 + r0;
    int p0 = pBase + cn;
    float f0 = (p0 < P_) ? xb[(2 * kp) * P_ + p0]     : 0.f;
    float f1 = (p0 < P_) ? xb[(2 * kp + 1) * P_ + p0] : 0.f;
    ldsB[cn * 132 + kp] = f2bf(f0) | (f2bf(f1) << 16);
  }
  __syncthreads();

  const int lane = tid & 31, w = tid >> 5;
  v8f acc[2][4];
#pragma unroll
  for (int mi = 0; mi < 2; ++mi)
#pragma unroll
    for (int ni = 0; ni < 4; ++ni)
#pragma unroll
      for (int q = 0; q < 8; ++q) acc[mi][ni][q] = 0.f;

  union Frag { u32x4 q[2]; v16bf v; };
#pragma unroll
  for (int ks = 0; ks < 8; ++ks) {
    Frag a[2];
#pragma unroll
    for (int mi = 0; mi < 2; ++mi) {
      const u32x4* ap = (const u32x4*)(Aswz + ((((2 * w + mi) * 8) + ks) * 32 + lane) * 8);
      a[mi].q[0] = ap[0]; a[mi].q[1] = ap[1];
    }
#pragma unroll
    for (int ni = 0; ni < 4; ++ni) {
      Frag bf;
      int off = (ni * 16 + (lane & 15)) * 132 + ks * 16 + 4 * (lane >> 4);
      bf.q[0] = *(const u32x4*)&ldsB[off];
      bf.q[1] = *(const u32x4*)&ldsB[off + 8];
#pragma unroll
      for (int mi = 0; mi < 2; ++mi)
        acc[mi][ni] = __builtin_amdgcn_wmma_f32_16x16x32_bf16(
            false, a[mi].v, false, bf.v, (short)0, acc[mi][ni], false, false);
    }
  }

  // epilogue (C/D layout: VGPR i -> M=i+8*half, N=lane&15)
  const int half = lane >> 4, nn = lane & 15;
#pragma unroll
  for (int mi = 0; mi < 2; ++mi) {
    int obase = (2 * w + mi) * 16 + 8 * half;
#pragma unroll
    for (int ni = 0; ni < 4; ++ni) {
      int p = pBase + ni * 16 + nn;
      if (p < P_) {
        float dd = ddt[ni * 16 + nn];
#pragma unroll
        for (int i = 0; i < 8; ++i) {
          int o = obase + i;
          size_t idx = ((size_t)b * C_ + o) * P_ + p;
          f32x4 e = ep[o];
          out[idx] = (acc[mi][ni][i] * e.x + e.y) * (e.z + dd) + x[idx];
        }
      }
    }
  }
}

// ---------------------------------------------------------------------------
extern "C" void kernel_launch(void* const* d_in, const int* in_sizes, int n_in,
                              void* d_out, int out_size, void* d_ws, size_t ws_size,
                              hipStream_t stream) {
  (void)in_sizes; (void)n_in; (void)out_size; (void)ws_size;
  const float* x     = (const float*)d_in[0];
  const float* qc_w  = (const float*)d_in[1];
  const float* qc_bn = (const float*)d_in[2];
  const float* kc_w  = (const float*)d_in[3];
  const float* kc_bn = (const float*)d_in[4];
  const float* lc_bn = (const float*)d_in[5];
  const float* qf_w  = (const float*)d_in[6];
  const float* qf_bn = (const float*)d_in[7];
  const float* kf_w  = (const float*)d_in[8];
  const float* kf_bn = (const float*)d_in[9];
  const float* lf_bn = (const float*)d_in[10];
  const float* qt_w  = (const float*)d_in[11];
  const float* qt_bn = (const float*)d_in[12];
  const float* kt_w  = (const float*)d_in[13];
  const float* kt_bn = (const float*)d_in[14];
  const float* lt_bn = (const float*)d_in[15];
  const float* v_w   = (const float*)d_in[16];
  const float* v_b   = (const float*)d_in[17];
  const float* v_bn  = (const float*)d_in[18];
  float* ws  = (float*)d_ws;
  float* out = (float*)d_out;

  prep_eff<<<6, 256, 0, stream>>>(qc_w, qc_bn, kc_w, kc_bn, qf_w, qf_bn,
                                  kf_w, kf_bn, qt_w, qt_bn, kt_w, kt_bn, ws);
  prep_misc<<<133, 256, 0, stream>>>(v_w, v_b, v_bn, lc_bn, lf_bn, lt_bn, ws);
  qv_pass<<<NF, 256, 0, stream>>>(x, ws);
  diag_cf<<<NF, 256, 0, stream>>>(ws + WS_QVC, ws + WS_KVC, 256, ws + WS_SCAL + 8, ws + WS_DC);
  diag_cf<<<NF, 256, 0, stream>>>(ws + WS_QVF, ws + WS_KVF, 251, ws + WS_SCAL + 9, ws + WS_DF);
  diag_t<<<(NT + 255) / 256, 256, 0, stream>>>(ws, ws + WS_DT);
  dim3 grid(28, B_);  // 28*64 = 1792 >= 1757 columns
  main_gemm<<<grid, 256, 0, stream>>>(x, ws, out);
}
